// Transformer_38577396252701
// MI455X (gfx1250) — matrix operand order repaired and checked by reading, not compiled, not run on hardware
//
#include <hip/hip_runtime.h>
#include <math.h>

// ---------------- problem constants ----------------
constexpr int Bc = 4, LQc = 512, LKc = 1024, Dc = 512, Hc = 8, DFFc = 2048, DHc = 64;

// ---------------- vector types ----------------
typedef __attribute__((ext_vector_type(16))) __bf16 v16bf;
typedef __attribute__((ext_vector_type(8)))  float  v8f;
typedef __attribute__((ext_vector_type(4)))  unsigned uint4v;
typedef __attribute__((ext_vector_type(4)))  float  float4v;

struct BF16x16 { uint4v lo, hi; };
static_assert(sizeof(BF16x16) == sizeof(v16bf), "bitcast size");

#define TS 40  // LDS row stride in bf16 elems (80B: 16B-aligned, bank-spread)

// ---------------- helpers ----------------
__device__ __forceinline__ unsigned bf1(float f) {   // rne bf16 in high 16 bits (unshifted pos kept high)
  unsigned u = __float_as_uint(f);
  return u + 0x7FFFu + ((u >> 16) & 1u);
}
__device__ __forceinline__ unsigned pack2bf(float lo, float hi) {
  return (bf1(lo) >> 16) | (bf1(hi) & 0xFFFF0000u);
}
__device__ __forceinline__ unsigned short f2bf(float f) { return (unsigned short)(bf1(f) >> 16); }
__device__ __forceinline__ unsigned f2ord(float f) {   // monotone float->uint
  unsigned u = __float_as_uint(f);
  return (u & 0x80000000u) ? ~u : (u | 0x80000000u);
}
__device__ __forceinline__ float ord2f(unsigned s) {
  unsigned u = (s & 0x80000000u) ? (s & 0x7FFFFFFFu) : ~s;
  return __uint_as_float(u);
}
// A fragment: lane = row; elems = K[kb..kb+15], kb in {0,16}
__device__ __forceinline__ v16bf ldfragA(const unsigned short* row, int kb) {
  BF16x16 t;
  t.lo = *(const uint4v*)(row + kb);
  t.hi = *(const uint4v*)(row + kb + 8);
  return __builtin_bit_cast(v16bf, t);
}
// B fragment: lane = col n (k-contiguous row in LDS); elems = K[kb..kb+15], kb in {0,16}
__device__ __forceinline__ v16bf ldfragB(const unsigned short* row, int kb) {
  BF16x16 t;
  t.lo = *(const uint4v*)(row + kb);
  t.hi = *(const uint4v*)(row + kb + 8);
  return __builtin_bit_cast(v16bf, t);
}
#define WMMA_BF16(A, Bm, C) \
  __builtin_amdgcn_wmma_f32_16x16x32_bf16(false, (A), false, (Bm), (short)0, (C), false, false)

// ============================================================
// Generic GEMM: C[M,N] = act(A[M,K] @ W[K,N] + bias[N])
// block tile 128x128x32, 256 threads (8 wave32), wave -> 32x64 (2x4 WMMA tiles)
// ============================================================
__global__ __launch_bounds__(256)
void gemm_bias_kernel(const float* __restrict__ A, const float* __restrict__ W,
                      const float* __restrict__ bias, float* __restrict__ C,
                      int N, int K, int relu) {
  __shared__ unsigned short sA[128 * TS];
  __shared__ unsigned short sW[128 * TS];
  const int tid = threadIdx.x, lane = tid & 31, wave = tid >> 5;
  const int wm = wave & 3, wn = wave >> 2;
  const int half = lane >> 4, l16 = lane & 15;
  const int m0 = blockIdx.y * 128, n0 = blockIdx.x * 128;
  const int akb = half ? 16 : 0, bkb = half ? 16 : 0;

  v8f acc[2][4] = {};
  for (int k0 = 0; k0 < K; k0 += 32) {
    if (k0 + 32 < K)   // hint next A k-slab into cache (global_prefetch_b8)
      __builtin_prefetch(A + (size_t)(m0 + (tid >> 1)) * K + k0 + 32, 0, 1);
    // ---- A tile: 128x32, 8 contiguous floats per segment, 2 segs/thread ----
    #pragma unroll
    for (int it = 0; it < 2; ++it) {
      int s = tid + it * 256;           // 0..511
      int r = s >> 2, cs = (s & 3) << 3;
      const float4v* gp = (const float4v*)(A + (size_t)(m0 + r) * K + k0 + cs);
      float4v f0 = gp[0], f1 = gp[1];
      uint4v pv = { pack2bf(f0.x, f0.y), pack2bf(f0.z, f0.w),
                    pack2bf(f1.x, f1.y), pack2bf(f1.z, f1.w) };
      *(uint4v*)(sA + r * TS + cs) = pv;
    }
    // ---- W tile: 32x128 -> transposed [n][k]; pack k-pairs, 2 segs/thread ----
    #pragma unroll
    for (int it = 0; it < 2; ++it) {
      int s = tid + it * 256;           // 0..511
      int kp2 = s >> 5;                 // 0..15 (k pair)
      int ns  = (s & 31) << 2;          // 0..124
      const float* w0 = W + (size_t)(k0 + 2 * kp2) * N + n0 + ns;
      float4v fa = *(const float4v*)w0;
      float4v fb = *(const float4v*)(w0 + N);
      *(unsigned*)(sW + (ns + 0) * TS + 2 * kp2) = pack2bf(fa.x, fb.x);
      *(unsigned*)(sW + (ns + 1) * TS + 2 * kp2) = pack2bf(fa.y, fb.y);
      *(unsigned*)(sW + (ns + 2) * TS + 2 * kp2) = pack2bf(fa.z, fb.z);
      *(unsigned*)(sW + (ns + 3) * TS + 2 * kp2) = pack2bf(fa.w, fb.w);
    }
    __syncthreads();
    v16bf af[2], bfv[4];
    #pragma unroll
    for (int tm = 0; tm < 2; ++tm)
      af[tm] = ldfragA(&sA[(wm * 32 + tm * 16 + l16) * TS], akb);
    #pragma unroll
    for (int tn = 0; tn < 4; ++tn)
      bfv[tn] = ldfragB(&sW[(wn * 64 + tn * 16 + l16) * TS], bkb);
    #pragma unroll
    for (int tm = 0; tm < 2; ++tm)
      #pragma unroll
      for (int tn = 0; tn < 4; ++tn)
        acc[tm][tn] = WMMA_BF16(af[tm], bfv[tn], acc[tm][tn]);
    __syncthreads();
  }
  #pragma unroll
  for (int tm = 0; tm < 2; ++tm)
    #pragma unroll
    for (int tn = 0; tn < 4; ++tn) {
      int n  = n0 + wn * 64 + tn * 16 + l16;
      int mb = m0 + wm * 32 + tm * 16 + half * 8;
      float bv = bias[n];
      #pragma unroll
      for (int r = 0; r < 8; ++r) {
        float v = acc[tm][tn][r] + bv;
        if (relu) v = fmaxf(v, 0.0f);
        C[(size_t)(mb + r) * N + n] = v;
      }
    }
}

// ============================================================
// logits[b,h,k,q] = (K_h@Q_h^T/8 - 1e9*mask)*aug_at + aug_hb*thb + aug_pi*tpi
// block tile 128(LK) x 128(LQ), K=DH=64; per-(b,h) ordered-uint atomicMax
// ============================================================
__global__ __launch_bounds__(256)
void logits_kernel(const float* __restrict__ kp, const float* __restrict__ qp,
                   const float* __restrict__ mask, const float* __restrict__ thb,
                   const float* __restrict__ tpi,
                   const float* __restrict__ aug_hb, const float* __restrict__ aug_pi,
                   const float* __restrict__ aug_at,
                   float* __restrict__ logits, unsigned* __restrict__ maxbuf) {
  __shared__ unsigned short sA[128 * TS];
  __shared__ unsigned short sW[128 * TS];
  __shared__ float red[256];
  const int tid = threadIdx.x, lane = tid & 31, wave = tid >> 5;
  const int wm = wave & 3, wn = wave >> 2;
  const int half = lane >> 4, l16 = lane & 15;
  const int z = blockIdx.z, b = z >> 3, h = z & 7;
  const int m0 = blockIdx.y * 128;   // LK dim
  const int n0 = blockIdx.x * 128;   // LQ dim
  const int akb = half ? 16 : 0, bkb = half ? 16 : 0;
  const float* Ab = kp + (size_t)b * LKc * Dc + h * DHc;   // rows: k, stride D
  const float* Bb = qp + (size_t)b * LQc * Dc + h * DHc;   // rows: q, stride D

  v8f acc[2][4] = {};
  for (int k0 = 0; k0 < DHc; k0 += 32) {
    #pragma unroll
    for (int it = 0; it < 2; ++it) {
      int s = tid + it * 256;
      int r = s >> 2, cs = (s & 3) << 3;
      const float4v* gp = (const float4v*)(Ab + (size_t)(m0 + r) * Dc + k0 + cs);
      float4v f0 = gp[0], f1 = gp[1];
      uint4v pv = { pack2bf(f0.x, f0.y), pack2bf(f0.z, f0.w),
                    pack2bf(f1.x, f1.y), pack2bf(f1.z, f1.w) };
      *(uint4v*)(sA + r * TS + cs) = pv;
    }
    #pragma unroll
    for (int it = 0; it < 2; ++it) {   // Q rows are k-contiguous: same pattern
      int s = tid + it * 256;
      int r = s >> 2, cs = (s & 3) << 3;
      const float4v* gp = (const float4v*)(Bb + (size_t)(n0 + r) * Dc + k0 + cs);
      float4v f0 = gp[0], f1 = gp[1];
      uint4v pv = { pack2bf(f0.x, f0.y), pack2bf(f0.z, f0.w),
                    pack2bf(f1.x, f1.y), pack2bf(f1.z, f1.w) };
      *(uint4v*)(sW + r * TS + cs) = pv;
    }
    __syncthreads();
    v16bf af[2], bfv[4];
    #pragma unroll
    for (int tm = 0; tm < 2; ++tm)
      af[tm] = ldfragA(&sA[(wm * 32 + tm * 16 + l16) * TS], akb);
    #pragma unroll
    for (int tn = 0; tn < 4; ++tn)
      bfv[tn] = ldfragB(&sW[(wn * 64 + tn * 16 + l16) * TS], bkb);
    #pragma unroll
    for (int tm = 0; tm < 2; ++tm)
      #pragma unroll
      for (int tn = 0; tn < 4; ++tn)
        acc[tm][tn] = WMMA_BF16(af[tm], bfv[tn], acc[tm][tn]);
    __syncthreads();
  }
  const float aat = aug_at[0], ahb = aug_hb[0], api = aug_pi[0];
  const size_t mtb = (size_t)b * LKc * LQc;   // mask/tables: (B,1,LK,LQ)
  const size_t lb  = (size_t)z * LKc * LQc;
  float lmax = -__builtin_inff();
  #pragma unroll
  for (int tm = 0; tm < 2; ++tm)
    #pragma unroll
    for (int tn = 0; tn < 4; ++tn) {
      int n  = n0 + wn * 64 + tn * 16 + l16;
      int mb = m0 + wm * 32 + tm * 16 + half * 8;
      #pragma unroll
      for (int r = 0; r < 8; ++r) {
        size_t off = (size_t)(mb + r) * LQc + n;
        float l = acc[tm][tn][r] * 0.125f;                 // 1/sqrt(64)
        l = (l - 1e9f * mask[mtb + off]) * aat + ahb * thb[mtb + off] + api * tpi[mtb + off];
        logits[lb + off] = l;
        lmax = fmaxf(lmax, l);
      }
    }
  red[tid] = lmax; __syncthreads();
  for (int s = 128; s > 0; s >>= 1) {
    if (tid < s) red[tid] = fmaxf(red[tid], red[tid + s]);
    __syncthreads();
  }
  if (tid == 0) atomicMax(&maxbuf[z], f2ord(red[0]));
}

// ============================================================
// in-place e = exp(l - max[z]); partial sums -> sumbuf[z]  (b128 per thread)
// ============================================================
__global__ __launch_bounds__(256)
void softmax_exp_kernel(float* __restrict__ logits, const unsigned* __restrict__ maxbuf,
                        float* __restrict__ sumbuf) {
  __shared__ float red[256];
  const int z = blockIdx.y;
  const float mx = ord2f(maxbuf[z]);
  const size_t i = (size_t)z * LKc * LQc + ((size_t)blockIdx.x * 256 + threadIdx.x) * 4;
  float4v v = *(float4v*)(logits + i);
  v.x = __expf(v.x - mx); v.y = __expf(v.y - mx);
  v.z = __expf(v.z - mx); v.w = __expf(v.w - mx);
  *(float4v*)(logits + i) = v;
  red[threadIdx.x] = v.x + v.y + v.z + v.w; __syncthreads();
  for (int st = 128; st > 0; st >>= 1) {
    if (threadIdx.x < st) red[threadIdx.x] += red[threadIdx.x + st];
    __syncthreads();
  }
  if (threadIdx.x == 0) atomicAdd(&sumbuf[z], red[0]);
}

// ============================================================
// out[b,q,h*64+d] = sum_k (e[z,k,q]*LK/sum[z]) * V[b,k,h*64+d]
// tile 128(q) x 64(d) x 32(k); scale fused into bf16 staging
// ============================================================
__global__ __launch_bounds__(256)
void attnv_kernel(const float* __restrict__ e, const float* __restrict__ vp,
                  const float* __restrict__ sumbuf, float* __restrict__ concat) {
  __shared__ unsigned short sA[128 * TS];
  __shared__ unsigned short sW[64 * TS];
  const int tid = threadIdx.x, lane = tid & 31, wave = tid >> 5;
  const int wm = wave & 3, wn = wave >> 2;
  const int half = lane >> 4, l16 = lane & 15;
  const int z = blockIdx.z, b = z >> 3, h = z & 7;
  const int q0 = blockIdx.y * 128;
  const int akb = half ? 16 : 0, bkb = half ? 16 : 0;
  const float scale = (float)LKc / sumbuf[z];
  const size_t eb = (size_t)z * LKc * LQc;
  const float* Vb = vp + (size_t)b * LKc * Dc + h * DHc;

  v8f acc[2][2] = {};
  for (int k0 = 0; k0 < LKc; k0 += 32) {
    // ---- e^T tile: read [k][q] (q-contiguous), pack k-pairs -> [q][k] ----
    #pragma unroll
    for (int it = 0; it < 2; ++it) {
      int s = tid + it * 256;           // 0..511
      int kp2 = s >> 5;                 // 0..15
      int qs  = (s & 31) << 2;          // 0..124
      const float* e0 = e + eb + (size_t)(k0 + 2 * kp2) * LQc + q0 + qs;
      float4v fa = *(const float4v*)e0;
      float4v fb = *(const float4v*)(e0 + LQc);
      *(unsigned*)(sA + (qs + 0) * TS + 2 * kp2) = pack2bf(fa.x * scale, fb.x * scale);
      *(unsigned*)(sA + (qs + 1) * TS + 2 * kp2) = pack2bf(fa.y * scale, fb.y * scale);
      *(unsigned*)(sA + (qs + 2) * TS + 2 * kp2) = pack2bf(fa.z * scale, fb.z * scale);
      *(unsigned*)(sA + (qs + 3) * TS + 2 * kp2) = pack2bf(fa.w * scale, fb.w * scale);
    }
    // ---- V tile: read [k][d] (d-contiguous), pack k-pairs -> [d][k] ----
    {
      int kp2 = tid >> 4;               // 0..15
      int ds  = (tid & 15) << 2;        // 0..60
      const float* v0 = Vb + (size_t)(k0 + 2 * kp2) * Dc + ds;
      float4v fa = *(const float4v*)v0;
      float4v fb = *(const float4v*)(v0 + Dc);
      *(unsigned*)(sW + (ds + 0) * TS + 2 * kp2) = pack2bf(fa.x, fb.x);
      *(unsigned*)(sW + (ds + 1) * TS + 2 * kp2) = pack2bf(fa.y, fb.y);
      *(unsigned*)(sW + (ds + 2) * TS + 2 * kp2) = pack2bf(fa.z, fb.z);
      *(unsigned*)(sW + (ds + 3) * TS + 2 * kp2) = pack2bf(fa.w, fb.w);
    }
    __syncthreads();
    v16bf a0 = ldfragA(&sA[(wm * 32 +      l16) * TS], akb);
    v16bf a1 = ldfragA(&sA[(wm * 32 + 16 + l16) * TS], akb);
    v16bf b0 = ldfragB(&sW[(wn * 32 +      l16) * TS], bkb);
    v16bf b1 = ldfragB(&sW[(wn * 32 + 16 + l16) * TS], bkb);
    acc[0][0] = WMMA_BF16(a0, b0, acc[0][0]);
    acc[0][1] = WMMA_BF16(a0, b1, acc[0][1]);
    acc[1][0] = WMMA_BF16(a1, b0, acc[1][0]);
    acc[1][1] = WMMA_BF16(a1, b1, acc[1][1]);
    __syncthreads();
  }
  #pragma unroll
  for (int tm = 0; tm < 2; ++tm)
    #pragma unroll
    for (int tn = 0; tn < 2; ++tn) {
      int n  = wn * 32 + tn * 16 + l16;                    // d within head
      int mb = q0 + wm * 32 + tm * 16 + half * 8;          // q
      #pragma unroll
      for (int r = 0; r < 8; ++r)
        concat[(size_t)b * LQc * Dc + (size_t)(mb + r) * Dc + h * DHc + n] = acc[tm][tn][r];
    }
}

// ============================================================
// out = LayerNorm(x + res) * g + b   (D=512, one 128-thread block per row, b128 I/O)
// ============================================================
__global__ __launch_bounds__(128)
void add_ln_kernel(const float* __restrict__ x, const float* __restrict__ res,
                   const float* __restrict__ g, const float* __restrict__ bb,
                   float* __restrict__ out) {
  __shared__ float rs[128], rq[128];
  const int row = blockIdx.x, tid = threadIdx.x;
  const size_t base = (size_t)row * Dc + tid * 4;
  float4v v = *(const float4v*)(x + base);
  float4v r = *(const float4v*)(res + base);
  v = v + r;
  rs[tid] = v.x + v.y + v.z + v.w;
  rq[tid] = v.x * v.x + v.y * v.y + v.z * v.z + v.w * v.w;
  __syncthreads();
  for (int s = 64; s > 0; s >>= 1) {
    if (tid < s) { rs[tid] += rs[tid + s]; rq[tid] += rq[tid + s]; }
    __syncthreads();
  }
  float mu  = rs[0] * (1.0f / 512.0f);
  float var = rq[0] * (1.0f / 512.0f) - mu * mu;
  float inv = rsqrtf(var + 1e-9f);
  float4v gv = *(const float4v*)(g + tid * 4);
  float4v bv = *(const float4v*)(bb + tid * 4);
  float4v o;
  o.x = (v.x - mu) * inv * gv.x + bv.x;
  o.y = (v.y - mu) * inv * gv.y + bv.y;
  o.z = (v.z - mu) * inv * gv.z + bv.z;
  o.w = (v.w - mu) * inv * gv.w + bv.w;
  *(float4v*)(out + base) = o;
}

__global__ void init_kernel(unsigned* __restrict__ maxbuf, float* __restrict__ sumbuf) {
  int i = threadIdx.x;
  if (i < Bc * Hc) { maxbuf[i] = f2ord(-__builtin_inff()); sumbuf[i] = 0.0f; }
}

// ============================================================
// host launcher
// ============================================================
extern "C" void kernel_launch(void* const* d_in, const int* in_sizes, int n_in,
                              void* d_out, int out_size, void* d_ws, size_t ws_size,
                              hipStream_t stream) {
  const float* q    = (const float*)d_in[0];
  const float* kv   = (const float*)d_in[1];
  const float* mask = (const float*)d_in[2];
  const float* thb  = (const float*)d_in[3];
  const float* tpi  = (const float*)d_in[4];
  const float* ahb  = (const float*)d_in[5];
  const float* api  = (const float*)d_in[6];
  const float* aat  = (const float*)d_in[7];
  const float* wq = (const float*)d_in[8];   const float* bq = (const float*)d_in[9];
  const float* wk = (const float*)d_in[10];  const float* bk = (const float*)d_in[11];
  const float* wv = (const float*)d_in[12];  const float* bv = (const float*)d_in[13];
  const float* wo = (const float*)d_in[14];  const float* bo = (const float*)d_in[15];
  const float* w1 = (const float*)d_in[16];  const float* b1 = (const float*)d_in[17];
  const float* w2 = (const float*)d_in[18];  const float* b2 = (const float*)d_in[19];
  const float* g1 = (const float*)d_in[20];  const float* be1 = (const float*)d_in[21];
  const float* g2 = (const float*)d_in[22];  const float* be2 = (const float*)d_in[23];

  float* ws = (float*)d_ws;
  size_t o = 0;
  float* qp = ws + o; o += (size_t)Bc * LQc * Dc;            // 4 MB
  float* kp = ws + o; o += (size_t)Bc * LKc * Dc;            // 8 MB
  float* vp = ws + o; o += (size_t)Bc * LKc * Dc;            // 8 MB
  float* lg = ws + o; o += (size_t)Bc * Hc * LKc * LQc;      // 64 MB
  unsigned* maxbuf = (unsigned*)(ws + o); o += 32;
  float* sumbuf = ws + o; o += 32;
  float* cc = ws + o; o += (size_t)Bc * LQc * Dc;            // 4 MB
  float* ao = ws + o; o += (size_t)Bc * LQc * Dc;            // 4 MB
  float* o1 = ws + o; o += (size_t)Bc * LQc * Dc;            // 4 MB
  float* f1 = lg;   // 16 MB, reuse dead logits region
  float* f2 = kp;   // 4 MB,  reuse dead K-proj region

  dim3 blk(256);
  init_kernel<<<1, 64, 0, stream>>>(maxbuf, sumbuf);
  gemm_bias_kernel<<<dim3(Dc / 128, (Bc * LQc) / 128), blk, 0, stream>>>(q,  wq, bq, qp, Dc, Dc, 0);
  gemm_bias_kernel<<<dim3(Dc / 128, (Bc * LKc) / 128), blk, 0, stream>>>(kv, wk, bk, kp, Dc, Dc, 0);
  gemm_bias_kernel<<<dim3(Dc / 128, (Bc * LKc) / 128), blk, 0, stream>>>(kv, wv, bv, vp, Dc, Dc, 0);
  logits_kernel<<<dim3(LQc / 128, LKc / 128, Bc * Hc), blk, 0, stream>>>(kp, qp, mask, thb, tpi, ahb, api, aat, lg, maxbuf);
  softmax_exp_kernel<<<dim3((LKc * LQc) / 1024, Bc * Hc), blk, 0, stream>>>(lg, maxbuf, sumbuf);
  attnv_kernel<<<dim3(1, LQc / 128, Bc * Hc), blk, 0, stream>>>(lg, vp, sumbuf, cc);
  gemm_bias_kernel<<<dim3(Dc / 128, (Bc * LQc) / 128), blk, 0, stream>>>(cc, wo, bo, ao, Dc, Dc, 0);
  add_ln_kernel<<<Bc * LQc, 128, 0, stream>>>(q, ao, g1, be1, o1);
  gemm_bias_kernel<<<dim3(DFFc / 128, (Bc * LQc) / 128), blk, 0, stream>>>(o1, w1, b1, f1, DFFc, Dc, 1);
  gemm_bias_kernel<<<dim3(Dc / 128, (Bc * LQc) / 128), blk, 0, stream>>>(f1, w2, b2, f2, Dc, DFFc, 0);
  add_ln_kernel<<<Bc * LQc, 128, 0, stream>>>(o1, f2, g2, be2, (float*)d_out);

  (void)in_sizes; (void)n_in; (void)out_size; (void)ws_size;
}
